// GatedGraphConvolution_34754875359431
// MI455X (gfx1250) — matrix operand order repaired and checked by reading, not compiled, not run on hardware
//
#include <hip/hip_runtime.h>

typedef __attribute__((ext_vector_type(16))) __bf16 v16bf;
typedef __attribute__((ext_vector_type(8)))  float  v8f;
typedef __attribute__((ext_vector_type(4)))  float  vf4;

#define DIN  64
#define DOUT 128

// ---------------------------------------------------------------------------
// Kernel 1: out = x  (residual clone), float4-vectorized
// ---------------------------------------------------------------------------
__global__ __launch_bounds__(256) void ggc_copy(const float* __restrict__ x,
                                                float* __restrict__ out, int n4) {
  int i = blockIdx.x * 256 + threadIdx.x;
  if (i < n4) ((vf4*)out)[i] = ((const vf4*)x)[i];
}

// ---------------------------------------------------------------------------
// Kernel 2: Y[n, 0:128] = x[n, :] @ W^T   (W: [128,64] row-major)
// bf16 WMMA, f32 accumulate. One wave -> one 16-row M-tile x all 128 columns.
// K=64 handled as two K=32 WMMA ops per N-tile.
//
// A (16x32 bf16) lane layout: lane l, h=l>>4, r=l&15 holds row M=r,
//   elems 0..7  -> K = 32*kb + 8h + i
//   elems 8..15 -> K = 32*kb + 16 + 8h + (i-8)
// B (32x16 bf16): same K order as A, N = r.
//   B[k, j] = W[j, k]  =>  W[j*64 + 32*kb + 8h + 0..7] and W[j*64 + 32*kb + 16 + 8h + 0..7].
// D (16x16 f32): acc[v] at lane -> row M = v + 8h, col N = r.
// ---------------------------------------------------------------------------
__global__ __launch_bounds__(128) void ggc_gemm(const float* __restrict__ x,
                                                const float* __restrict__ W,
                                                float* __restrict__ Y,
                                                int n_tiles, int n_nodes) {
  const int lane = threadIdx.x & 31;
  const int tile = blockIdx.x * 4 + (threadIdx.x >> 5);
  if (tile >= n_tiles) return;
  const int m0 = tile * 16;
  const int r  = lane & 15;
  const int h  = lane >> 4;

  int arow = m0 + r;
  if (arow >= n_nodes) arow = n_nodes - 1;   // clamp (tail stores are guarded)

  v16bf a[2];
  const float* xr = x + (long long)arow * DIN;
  #pragma unroll
  for (int kb = 0; kb < 2; ++kb) {
    const float* c0 = xr + kb * 32 + h * 8;
    const float* c1 = xr + kb * 32 + 16 + h * 8;
    #pragma unroll
    for (int i = 0; i < 8; ++i) a[kb][i]     = (__bf16)c0[i];
    #pragma unroll
    for (int i = 0; i < 8; ++i) a[kb][8 + i] = (__bf16)c1[i];
  }

  const bool full_tile = (m0 + 16 <= n_nodes);   // uniform across the wave
  // store base for this lane: row (m0 + 8h), column r
  float* yb = Y + (long long)(m0 + 8 * h) * DOUT + r;

  #pragma unroll
  for (int nt = 0; nt < 8; ++nt) {
    const int j = nt * 16 + r;               // output column == row of W
    v8f acc = {};
    #pragma unroll
    for (int kb = 0; kb < 2; ++kb) {
      const float* wp = W + j * DIN + kb * 32 + h * 8;
      v16bf b;
      #pragma unroll
      for (int i = 0; i < 16; ++i) b[i] = (__bf16)wp[i < 8 ? i : i + 8];
      acc = __builtin_amdgcn_wmma_f32_16x16x32_bf16(
          /*neg_a=*/false, a[kb], /*neg_b=*/false, b,
          /*c_mod=*/(short)0, acc, /*reuse_a=*/false, /*reuse_b=*/false);
    }
    if (full_tile) {
      // branch-free path: 8 coalesced 64B stores (16 lanes x 4B, column-contig)
      #pragma unroll
      for (int v = 0; v < 8; ++v)
        yb[(long long)v * DOUT + nt * 16] = acc[v];
    } else {
      #pragma unroll
      for (int v = 0; v < 8; ++v) {
        const int row = m0 + v + 8 * h;
        if (row < n_nodes)
          Y[(long long)row * DOUT + nt * 16 + r] = acc[v];
      }
    }
  }
}

// ---------------------------------------------------------------------------
// Kernel 3: per edge e: out[src[e], :] += sigmoid(Y[tgt[e], :64]) * Y[tgt[e], 64:]
// One wave per edge: each lane handles 2 feature elements (float2 of g and e),
// coalesced 512B read of Y (L2-resident), 64 contiguous f32 atomics per edge.
// ---------------------------------------------------------------------------
__global__ __launch_bounds__(256) void ggc_edge(const float* __restrict__ Y,
                                                const int* __restrict__ esrc,
                                                const int* __restrict__ etgt,
                                                float* __restrict__ out,
                                                int n_edges) {
  long long gid = (long long)blockIdx.x * 256 + threadIdx.x;
  int e = (int)(gid >> 5);
  if (e >= n_edges) return;
  int l = ((int)gid) & 31;

  int t = etgt[e];
  int s = esrc[e];

  const float* yb = Y + (long long)t * DOUT + l * 2;
  float g0 = yb[0],  g1 = yb[1];
  float e0 = yb[64], e1 = yb[65];

  float m0 = e0 * (1.0f / (1.0f + __expf(-g0)));
  float m1 = e1 * (1.0f / (1.0f + __expf(-g1)));

  float* ob = out + (long long)s * DIN + l * 2;
  unsafeAtomicAdd(ob,     m0);   // global_atomic_add_f32
  unsafeAtomicAdd(ob + 1, m1);
}

// ---------------------------------------------------------------------------
extern "C" void kernel_launch(void* const* d_in, const int* in_sizes, int n_in,
                              void* d_out, int out_size, void* d_ws, size_t ws_size,
                              hipStream_t stream) {
  const float* x    = (const float*)d_in[0];
  const float* W    = (const float*)d_in[1];
  const int*   esrc = (const int*)d_in[2];
  const int*   etgt = (const int*)d_in[3];
  // d_in[4] (distance_nbr) unused by the reference path.

  float* out = (float*)d_out;
  float* Y   = (float*)d_ws;     // needs n_nodes*128*4 = 25.6 MB scratch

  const int n_nodes = in_sizes[0] / DIN;
  const int n_edges = in_sizes[2];

  // 1) residual copy
  const int n4 = (n_nodes * DIN) / 4;
  ggc_copy<<<(n4 + 255) / 256, 256, 0, stream>>>(x, out, n4);

  // 2) node-level GEMM via WMMA bf16
  const int n_tiles = (n_nodes + 15) / 16;
  ggc_gemm<<<(n_tiles + 3) / 4, 128, 0, stream>>>(x, W, Y, n_tiles, n_nodes);

  // 3) edge gather + gate + scatter-add
  const long long tot = (long long)n_edges * 32;
  const int blocks = (int)((tot + 255) / 256);
  ggc_edge<<<blocks, 256, 0, stream>>>(Y, esrc, etgt, out, n_edges);
}
